// LePEAttention_15753940041988
// MI455X (gfx1250) — matrix operand order repaired and checked by reading, not compiled, not run on hardware
//
#include <hip/hip_runtime.h>
#include <hip/hip_bf16.h>

typedef __attribute__((ext_vector_type(16))) _Float16 v16h;
typedef __attribute__((ext_vector_type(8)))  float    v8f;
typedef __attribute__((ext_vector_type(4)))  unsigned u32x4;
typedef __attribute__((ext_vector_type(8)))  int      i32x8;
typedef __attribute__((ext_vector_type(4)))  int      i32x4;

#define HEADS 12
#define HD    32
#define CH    384
#define HH    64
#define WW    64
#define BB    16
#define NN    (HH*WW)          // 4096
#define SS    (BB*WW)          // 1024 strips
#define LL    64               // strip length
#define QSZ   ((size_t)SS*HEADS*LL*HD)   // 25,165,824 halves per q/k/v
#define SCALE 0.17677669529663687f       // 32^-0.5

#if defined(__has_builtin)
#if __has_builtin(__builtin_amdgcn_tensor_load_to_lds) && \
    __has_builtin(__builtin_amdgcn_s_wait_tensorcnt)
#define USE_TDM 1
#endif
#endif

union FragU { v16h v; unsigned u[8]; };

// A-matrix fragment (16x32 f16, M x K). Per ISA 7.12.2:
// lane L -> M = L%16 ; halves e: K = e + (e>=8?8:0) + (L>=16?8:0)
__device__ __forceinline__ v16h load_a_frag(const _Float16* base, int stride,
                                            int row0, int kbase, int lane) {
  FragU f;
  const _Float16* rp = base + (size_t)(row0 + (lane & 15)) * stride + kbase + ((lane >> 4) << 3);
#pragma unroll
  for (int p = 0; p < 8; ++p)
    f.u[p] = *(const unsigned*)(rp + 2 * p + ((p >= 4) ? 8 : 0));
  return f.v;
}

// B-matrix fragment (32x16 f16, K x N) from LDS stored transposed Bt[n][k]:
// lane L -> N = L%16 ; halves e: K = e + (e>=8?8:0) + (L>=16?8:0)
__device__ __forceinline__ v16h load_b_frag(const _Float16* base, int stride,
                                            int col0, int kbase, int lane) {
  FragU f;
  const _Float16* rp = base + (size_t)(col0 + (lane & 15)) * stride + kbase + ((lane >> 4) << 3);
#pragma unroll
  for (int p = 0; p < 8; ++p)
    f.u[p] = *(const unsigned*)(rp + 2 * p + ((p >= 4) ? 8 : 0));
  return f.v;
}

__device__ __forceinline__ v8f zero8() {
  v8f z = {0.f, 0.f, 0.f, 0.f, 0.f, 0.f, 0.f, 0.f};
  return z;
}

#ifdef USE_TDM
// Issue one TDM 1-D tile copy: 2048 contiguous f16 (4 KB) global -> LDS.
// D# layout per CDNA5 ISA.
__device__ __forceinline__ void tdm_load_tile(unsigned lds_off, const _Float16* gsrc) {
  unsigned long long ga = (unsigned long long)(size_t)gsrc;
  u32x4 g0;
  g0[0] = 1u;                                   // count=1, user-mode descriptor
  g0[1] = lds_off;                              // lds_addr (bytes)
  g0[2] = (unsigned)(ga & 0xFFFFFFFFu);         // global_addr[31:0]
  g0[3] = (unsigned)((ga >> 32) & 0x01FFFFFFu)  // global_addr[56:32]
          | 0x80000000u;                        // type = 2 ("image")
  i32x8 g1;
  g1[0] = 0x00010000;          // wg_mask=0, data_size=1 (2 bytes), no flags
  g1[1] = (int)(2048u << 16);  // tensor_dim0 = 2048 (bits 79:48 low half)
  g1[2] = (int)(1u << 16);     // tensor_dim0 hi = 0, tensor_dim1 = 1
  g1[3] = (int)(2048u << 16);  // tensor_dim1 hi = 0, tile_dim0 = 2048
  g1[4] = 1;                   // tile_dim1 = 1, tile_dim2 = 0
  g1[5] = 2048;                // tensor_dim0_stride (low 32)
  g1[6] = 0;                   // stride hi, tensor_dim1_stride lo
  g1[7] = 0;
  i32x4 gz = {0, 0, 0, 0};
#if __has_include(<hip/amd_detail/amd_gfx1250_TDM.h>)
  i32x8 gz8 = {0, 0, 0, 0, 0, 0, 0, 0};
  __builtin_amdgcn_tensor_load_to_lds(g0, g1, gz, gz, gz8, 0);
#else
  __builtin_amdgcn_tensor_load_to_lds(g0, g1, gz, gz, 0);
#endif
}
#endif

// ---------------------------------------------------------------------------
// K1: QKV GEMM.  rows r = strip*64 + l (strip-transposed view of x),
// cols n in [0,1152) = qi*384 + head*32 + d.  f16 WMMA, f32 accum.
// Block tile 64x128, 4 waves, wave tile 32x64 -> 8 WMMAs / 6 frag loads / step.
// ---------------------------------------------------------------------------
__global__ void __launch_bounds__(128)
qkv_gemm_kernel(const float* __restrict__ x, const float* __restrict__ qkv_w,
                _Float16* __restrict__ qh, _Float16* __restrict__ kh,
                _Float16* __restrict__ vh) {
  __shared__ _Float16 As[64 * 36];    // A tile, row-major [m][k], padded
  __shared__ _Float16 Bt[128 * 36];   // B tile, transposed [n][k], padded
  const int tid = threadIdx.x;
  const int lane = tid & 31, wave = tid >> 5;
  const int wm = (wave >> 1) * 32;    // 2 wave-rows
  const int wn = (wave & 1) * 64;     // 2 wave-cols
  const int bx = blockIdx.x;          // n-tile (9)
  const int by = blockIdx.y;          // m-tile (1024)

  v8f acc[2][4];
#pragma unroll
  for (int i = 0; i < 2; ++i)
#pragma unroll
    for (int j = 0; j < 4; ++j) acc[i][j] = zero8();

  for (int kb = 0; kb < CH; kb += 32) {
    __syncthreads();
    // fill A: x with strip transpose, convert f32->f16
#pragma unroll
    for (int i = tid; i < 64 * 32; i += 128) {
      int m = i >> 5, k = i & 31;
      int r = by * 64 + m;
      int b = r >> 12, rem = r & 4095;
      int w = rem >> 6, l = rem & 63;
      float v = x[((size_t)b * NN + (size_t)l * WW + w) * CH + kb + k];
      As[m * 36 + k] = (_Float16)v;
    }
    // fill Bt: qkv_w[n][k] row-major -> Bt[n][k]
#pragma unroll
    for (int i = tid; i < 128 * 32; i += 128) {
      int n = i >> 5, k = i & 31;
      Bt[n * 36 + k] = (_Float16)qkv_w[(size_t)(bx * 128 + n) * CH + kb + k];
    }
    if (kb + 32 < CH)
      __builtin_prefetch(&qkv_w[(size_t)(bx * 128 + tid) * CH + kb + 32], 0, 1);
    __syncthreads();

    v16h bfr[4];
#pragma unroll
    for (int nt = 0; nt < 4; ++nt) bfr[nt] = load_b_frag(Bt, 36, wn + nt * 16, 0, lane);
#pragma unroll
    for (int mt = 0; mt < 2; ++mt) {
      v16h a = load_a_frag(As, 36, wm + mt * 16, 0, lane);
#pragma unroll
      for (int nt = 0; nt < 4; ++nt)
        acc[mt][nt] = __builtin_amdgcn_wmma_f32_16x16x32_f16(
            false, a, false, bfr[nt], (short)0, acc[mt][nt], false, false);
    }
  }

  // epilogue: scatter into Q/K/V f16 tiles  (128-wide tile never crosses qi)
#pragma unroll
  for (int mt = 0; mt < 2; ++mt) {
#pragma unroll
    for (int nt = 0; nt < 4; ++nt) {
      int ncol = bx * 128 + wn + nt * 16 + (lane & 15);
      int qi = ncol / CH;
      int head = (ncol % CH) >> 5;
      int d = ncol & 31;
      _Float16* dst = (qi == 0) ? qh : (qi == 1) ? kh : vh;
      float mul = (qi == 0) ? SCALE : 1.0f;
#pragma unroll
      for (int r = 0; r < 8; ++r) {
        int row = by * 64 + wm + mt * 16 + r + ((lane >> 4) << 3);
        int s = row >> 6, l = row & 63;
        size_t off = (((size_t)(s * HEADS + head)) * LL + l) * HD + d;
        dst[off] = (_Float16)(acc[mt][nt][r] * mul);
      }
    }
  }
}

// ---------------------------------------------------------------------------
// K2: block-diagonal attention. One block (2 waves) per (strip, head).
// Q/K tiles fetched via TDM tensor_load_to_lds when available.
// S = Q K^T (scale pre-folded), softmax rows in f32, O = P V.
// Writes f16 into pre[b, w*H + l, head*32 + d] (faithful transposed layout).
// ---------------------------------------------------------------------------
__global__ void __launch_bounds__(64)
attn_kernel(const _Float16* __restrict__ qh, const _Float16* __restrict__ kh,
            const _Float16* __restrict__ vh, _Float16* __restrict__ pre) {
  __shared__ _Float16 Qs[64 * 32];   // [l][d] contiguous (TDM blit target)
  __shared__ _Float16 Ks[64 * 32];   // [key][d] contiguous == Bt layout
  __shared__ _Float16 Vt[32 * 72];   // [d][key]  == Bt layout for PV
  __shared__ float    Af[64 * 65];   // attention scores f32
  __shared__ _Float16 Ah[64 * 72];   // softmax probs f16, row-major

  const int tid = threadIdx.x;
  const int lane = tid & 31, wave = tid >> 5;
  const int wbase = wave * 32;
  const int sh = blockIdx.x;                 // strip*HEADS + head
  const size_t base = (size_t)sh * (LL * HD);

#ifdef USE_TDM
  if (wave == 0) {
    tdm_load_tile((unsigned)(size_t)&Qs[0], qh + base);
    tdm_load_tile((unsigned)(size_t)&Ks[0], kh + base);
  }
  for (int i = tid; i < LL * HD; i += 64) {  // V needs transpose: stage manually
    int l = i >> 5, d = i & 31;
    Vt[d * 72 + l] = vh[base + i];
  }
  if (wave == 0) __builtin_amdgcn_s_wait_tensorcnt(0);
#else
  for (int i = tid; i < LL * HD; i += 64) {
    int l = i >> 5, d = i & 31;
    Qs[i] = qh[base + i];
    Ks[i] = kh[base + i];
    Vt[d * 72 + l] = vh[base + i];
  }
#endif
  __syncthreads();

  // S = Q K^T : M=64 (split by wave), N=64, K=32 (one WMMA per tile pair)
  {
    v16h bfr[4];
#pragma unroll
    for (int nt = 0; nt < 4; ++nt) bfr[nt] = load_b_frag(Ks, 32, nt * 16, 0, lane);
#pragma unroll
    for (int mt = 0; mt < 2; ++mt) {
      v16h a = load_a_frag(Qs, 32, wbase + mt * 16, 0, lane);
#pragma unroll
      for (int nt = 0; nt < 4; ++nt) {
        v8f c = zero8();
        c = __builtin_amdgcn_wmma_f32_16x16x32_f16(false, a, false, bfr[nt],
                                                   (short)0, c, false, false);
#pragma unroll
        for (int r = 0; r < 8; ++r) {
          int row = wbase + mt * 16 + r + ((lane >> 4) << 3);
          int col = nt * 16 + (lane & 15);
          Af[row * 65 + col] = c[r];
        }
      }
    }
  }
  __syncthreads();

  // softmax: one row per thread (64 threads, 64 rows)
  {
    float m = -INFINITY;
#pragma unroll 8
    for (int j = 0; j < 64; ++j) m = fmaxf(m, Af[tid * 65 + j]);
    float sum = 0.f;
#pragma unroll 8
    for (int j = 0; j < 64; ++j) {
      float e = __expf(Af[tid * 65 + j] - m);
      sum += e;
      Af[tid * 65 + j] = e;
    }
    float inv = 1.f / sum;
#pragma unroll 8
    for (int j = 0; j < 64; ++j)
      Ah[tid * 72 + j] = (_Float16)(Af[tid * 65 + j] * inv);
  }
  __syncthreads();

  // O = P V : M=64, K=64 (2 steps), N=32
  v8f oacc[2][2];
#pragma unroll
  for (int i = 0; i < 2; ++i)
#pragma unroll
    for (int j = 0; j < 2; ++j) oacc[i][j] = zero8();

#pragma unroll
  for (int kt = 0; kt < 2; ++kt) {
    v16h bfr[2];
#pragma unroll
    for (int nt = 0; nt < 2; ++nt) bfr[nt] = load_b_frag(Vt, 72, nt * 16, kt * 32, lane);
#pragma unroll
    for (int mt = 0; mt < 2; ++mt) {
      v16h a = load_a_frag(Ah, 72, wbase + mt * 16, kt * 32, lane);
#pragma unroll
      for (int nt = 0; nt < 2; ++nt)
        oacc[mt][nt] = __builtin_amdgcn_wmma_f32_16x16x32_f16(
            false, a, false, bfr[nt], (short)0, oacc[mt][nt], false, false);
    }
  }

  // write to pre[b, w*H + l, head*32 + d]
  const int s = sh / HEADS, head = sh % HEADS;
  const int b = s >> 6, w = s & 63;
#pragma unroll
  for (int mt = 0; mt < 2; ++mt)
#pragma unroll
    for (int nt = 0; nt < 2; ++nt)
#pragma unroll
      for (int r = 0; r < 8; ++r) {
        int l = wbase + mt * 16 + r + ((lane >> 4) << 3);
        int d = nt * 16 + (lane & 15);
        size_t off = (((size_t)b * NN) + (size_t)w * HH + l) * CH + head * HD + d;
        pre[off] = (_Float16)oacc[mt][nt][r];
      }
}

// ---------------------------------------------------------------------------
// K3: LePE depthwise 3x3 conv (+bias), accumulated into pre (f16).
// ---------------------------------------------------------------------------
__global__ void __launch_bounds__(256)
lepe_kernel(const float* __restrict__ x, const float* __restrict__ lw,
            const float* __restrict__ lb, _Float16* __restrict__ pre) {
  size_t idx = (size_t)blockIdx.x * 256 + threadIdx.x;   // B*N*C elements
  int c = (int)(idx % CH);
  size_t rest = idx / CH;
  int w = (int)(rest & 63);
  size_t r2 = rest >> 6;
  int h = (int)(r2 & 63);
  int b = (int)(r2 >> 6);
  float sum = lb[c];
#pragma unroll
  for (int ki = 0; ki < 3; ++ki) {
    int hy = h + ki - 1;
    if (hy < 0 || hy >= HH) continue;
#pragma unroll
    for (int kj = 0; kj < 3; ++kj) {
      int wx = w + kj - 1;
      if (wx < 0 || wx >= WW) continue;
      sum += x[((size_t)b * NN + (size_t)hy * WW + wx) * CH + c] * lw[c * 9 + ki * 3 + kj];
    }
  }
  pre[idx] = (_Float16)((float)pre[idx] + sum);
}

// ---------------------------------------------------------------------------
// K4: projection GEMM  out = pre @ proj_w^T + proj_b  (f16 WMMA, f32 out)
// Block tile 64x128, wave tile 32x64.
// ---------------------------------------------------------------------------
__global__ void __launch_bounds__(128)
proj_gemm_kernel(const _Float16* __restrict__ pre, const float* __restrict__ pw,
                 const float* __restrict__ pb, float* __restrict__ out) {
  __shared__ _Float16 As[64 * 36];
  __shared__ _Float16 Bt[128 * 36];
  const int tid = threadIdx.x;
  const int lane = tid & 31, wave = tid >> 5;
  const int wm = (wave >> 1) * 32, wn = (wave & 1) * 64;
  const int bx = blockIdx.x;   // n-tile (3)
  const int by = blockIdx.y;   // m-tile (1024)

  v8f acc[2][4];
#pragma unroll
  for (int i = 0; i < 2; ++i)
#pragma unroll
    for (int j = 0; j < 4; ++j) acc[i][j] = zero8();

  for (int kb = 0; kb < CH; kb += 32) {
    __syncthreads();
#pragma unroll
    for (int i = tid; i < 64 * 32; i += 128) {
      int m = i >> 5, k = i & 31;
      As[m * 36 + k] = pre[(size_t)(by * 64 + m) * CH + kb + k];
    }
#pragma unroll
    for (int i = tid; i < 128 * 32; i += 128) {
      int n = i >> 5, k = i & 31;
      Bt[n * 36 + k] = (_Float16)pw[(size_t)(bx * 128 + n) * CH + kb + k];
    }
    if (kb + 32 < CH)
      __builtin_prefetch(&pw[(size_t)(bx * 128 + tid) * CH + kb + 32], 0, 1);
    __syncthreads();

    v16h bfr[4];
#pragma unroll
    for (int nt = 0; nt < 4; ++nt) bfr[nt] = load_b_frag(Bt, 36, wn + nt * 16, 0, lane);
#pragma unroll
    for (int mt = 0; mt < 2; ++mt) {
      v16h a = load_a_frag(As, 36, wm + mt * 16, 0, lane);
#pragma unroll
      for (int nt = 0; nt < 4; ++nt)
        acc[mt][nt] = __builtin_amdgcn_wmma_f32_16x16x32_f16(
            false, a, false, bfr[nt], (short)0, acc[mt][nt], false, false);
    }
  }

#pragma unroll
  for (int mt = 0; mt < 2; ++mt)
#pragma unroll
    for (int nt = 0; nt < 4; ++nt) {
      int ncol = bx * 128 + wn + nt * 16 + (lane & 15);
      float bias = pb[ncol];
#pragma unroll
      for (int r = 0; r < 8; ++r) {
        int row = by * 64 + wm + mt * 16 + r + ((lane >> 4) << 3);
        out[(size_t)row * CH + ncol] = acc[mt][nt][r] + bias;
      }
    }
}

// ---------------------------------------------------------------------------
extern "C" void kernel_launch(void* const* d_in, const int* in_sizes, int n_in,
                              void* d_out, int out_size, void* d_ws, size_t ws_size,
                              hipStream_t stream) {
  const float* x      = (const float*)d_in[0];
  const float* qkv_w  = (const float*)d_in[1];
  const float* proj_w = (const float*)d_in[2];
  const float* proj_b = (const float*)d_in[3];
  const float* lepe_w = (const float*)d_in[4];
  const float* lepe_b = (const float*)d_in[5];
  (void)in_sizes; (void)n_in; (void)out_size; (void)ws_size;

  _Float16* qh  = (_Float16*)d_ws;
  _Float16* kh  = qh + QSZ;
  _Float16* vh  = kh + QSZ;
  _Float16* pre = vh + QSZ;

  // 1) QKV GEMM: M=65536 (strip-transposed rows), N=1152, K=384
  qkv_gemm_kernel<<<dim3(1152 / 128, (SS * LL) / 64), 128, 0, stream>>>(x, qkv_w, qh, kh, vh);
  // 2) block-diagonal attention: one block per (strip, head)
  attn_kernel<<<SS * HEADS, 64, 0, stream>>>(qh, kh, vh, pre);
  // 3) LePE depthwise conv accumulated into pre
  lepe_kernel<<<(BB * NN * CH) / 256, 256, 0, stream>>>(x, lepe_w, lepe_b, pre);
  // 4) projection GEMM: M=65536, N=384, K=384, fp32 out + bias
  proj_gemm_kernel<<<dim3(CH / 128, (BB * NN) / 64), 128, 0, stream>>>(pre, proj_w, proj_b,
                                                                       (float*)d_out);
}
